// AttentionMechanism_29970281792254
// MI455X (gfx1250) — matrix operand order repaired and checked by reading, not compiled, not run on hardware
//
#include <hip/hip_runtime.h>
#include <math.h>

// Problem constants (reference): B=64, R=196, F=2048, H=1024
#define KB_ 64
#define KR_ 196
#define KF_ 2048
#define KH_ 1024
#define KM_ (KB_ * KR_)   // 12544 = 49 * 256

// Tiling: 256 threads = 8 waves (wave32). Block tile 256(M) x 64(N), K step 32.
// Each wave computes a 32x64 strip -> 2x4 accumulators (v8f), 8 WMMAs/K-step.
#define TM_ 256
#define TN_ 64
#define TK_ 32
#define NT_ 256
// LDS halfword stride 40 -> 80B rows: every 16B segment aligned for b128 async
// stores, and 20-bank stride is conflict-free across 16 lanes.
#define LDA_ 40
#define LDB_ 40

typedef __attribute__((ext_vector_type(16))) __bf16        v16bf;
typedef __attribute__((ext_vector_type(8)))  float         v8f;
typedef __attribute__((ext_vector_type(8)))  unsigned int  v8u;

__device__ __forceinline__ unsigned short f2bf(float f) {
  unsigned int u = __float_as_uint(f);
  u += 0x7FFFu + ((u >> 16) & 1u);   // RNE (off the hot path)
  return (unsigned short)(u >> 16);
}
__device__ __forceinline__ float bf2f(unsigned short s) {
  return __uint_as_float(((unsigned int)s) << 16);
}
// Pack two f32 to packed bf16 (truncate) in ONE VALU op: out = {hi[31:16], lo[31:16]}
__device__ __forceinline__ unsigned int pack_bf16_trunc(float lo, float hi) {
  return __builtin_amdgcn_perm(__float_as_uint(hi), __float_as_uint(lo), 0x07060302u);
}
__device__ __forceinline__ unsigned int lds_addr32(const void* p) {
  return (unsigned int)(uintptr_t)p;   // LDS generic ptr: low 32 bits = LDS offset
}
// CDNA5 async global->LDS copy, 16B per lane, tracked by ASYNCcnt.
__device__ __forceinline__ void async_ld16(unsigned int lds, unsigned int goff,
                                           unsigned long long base) {
  asm volatile("global_load_async_to_lds_b128 %0, %1, %2"
               :: "v"(lds), "v"(goff), "s"(base) : "memory");
}
__device__ __forceinline__ void wait_async0() {
  asm volatile("s_wait_asynccnt 0x0" ::: "memory");
}

// ---------------------------------------------------------------------------
// f32 -> bf16 conversion (weights, once)
// ---------------------------------------------------------------------------
__global__ __launch_bounds__(256) void cvt_f32_bf16_kernel(
    const float* __restrict__ in, unsigned short* __restrict__ out, int n) {
  int i = blockIdx.x * 256 + threadIdx.x;
  if (i < n) out[i] = f2bf(in[i]);
}

// ---------------------------------------------------------------------------
// bf16 WMMA GEMM:  C[M,N] = A[M,K] * B^T,  Bbf row-major [N,K] (bf16 in ws)
// MODE 0: out_bf16 = tanh(acc + bias[n])                   (GEMM1 -> v1)
// MODE 1: out_f32  = tanh(acc + bias[n] + qt[m/R][n])      (GEMM2 -> h_a)
// MODE 2: out_f32  = acc + bias[n]                         (q_t GEMM)
// ABF=true : A is bf16 -> async global->LDS staging (no VGPR round trip)
// ABF=false: A is f32  -> load clause into regs, then v_perm pack + ds_store
// ---------------------------------------------------------------------------
template <int MODE, bool ABF>
__global__ __launch_bounds__(NT_) void wmma_gemm_kernel(
    const float* __restrict__ Af, const unsigned short* __restrict__ Abf,
    const unsigned short* __restrict__ Bbf,
    const float* __restrict__ bias, const float* __restrict__ qt,
    float* __restrict__ outF, unsigned short* __restrict__ outBF,
    int M, int K, int N) {
  __shared__ unsigned short lsA[2][TM_ * LDA_];  // [m][k] bf16, double buffered
  __shared__ unsigned short lsB[2][TN_ * LDB_];  // [n][k] bf16, double buffered

  const int tid   = threadIdx.x;
  const int lane  = tid & 31;
  const int wv    = tid >> 5;      // 0..7
  const int l15   = lane & 15;
  const int lhalf = lane >> 4;

  const int mBase = blockIdx.y * TM_;
  const int nBase = blockIdx.x * TN_;
  const unsigned long long aBase = (unsigned long long)Abf;
  const unsigned long long bBase = (unsigned long long)Bbf;

  v8f acc[2][4] = {};   // wave strip: 32(M) x 64(N)

  // ---- staging helpers -----------------------------------------------------
  auto stageA = [&](int buf, int kb) {
    if (ABF) {
      // async: 256 rows x 64B, 16B chunks -> 4 issues per thread
      #pragma unroll
      for (int i = 0; i < 4; ++i) {
        int c = i * 256 + tid;
        int m = c >> 2, seg = c & 3;
        unsigned int lds = lds_addr32(&lsA[buf][m * LDA_ + seg * 8]);
        unsigned int go  = ((unsigned int)(mBase + m) * (unsigned int)K + kb + seg * 8) * 2u;
        async_ld16(lds, go, aBase);
      }
    } else {
      // Phase 1: issue ALL global b128 loads into registers (no early waits),
      // Phase 2: pack (1 v_perm per pair) + ds_store_b64. 8 chunks of 4 floats.
      float4 vb[8];
      #pragma unroll
      for (int i = 0; i < 8; ++i) {
        int c = i * 256 + tid;
        int m = c >> 3, q = c & 7;            // q: 4-float group within 32-k row
        int gm = mBase + m; if (gm >= M) gm = M - 1;
        vb[i] = *(const float4*)(Af + (size_t)gm * K + kb + 4 * q);
      }
      #pragma unroll
      for (int i = 0; i < 8; ++i) {
        int c = i * 256 + tid;
        int m = c >> 3, q = c & 7;
        uint2 pk;
        pk.x = pack_bf16_trunc(vb[i].x, vb[i].y);
        pk.y = pack_bf16_trunc(vb[i].z, vb[i].w);
        *(uint2*)&lsA[buf][m * LDA_ + 4 * q] = pk;   // (m*40+4q)*2 -> 8B aligned
      }
    }
  };
  auto stageB = [&](int buf, int kb) {
    // async: 64 rows x 64B, 16B chunks -> 1 issue per thread
    int n = tid >> 2, seg = tid & 3;
    unsigned int lds = lds_addr32(&lsB[buf][n * LDB_ + seg * 8]);
    unsigned int go  = ((unsigned int)(nBase + n) * (unsigned int)K + kb + seg * 8) * 2u;
    async_ld16(lds, go, bBase);
  };

  const int nK = K / TK_;
  stageA(0, 0);
  stageB(0, 0);

  int buf = 0;
  for (int it = 0; it < nK; ++it) {
    wait_async0();       // this wave's async stores into `buf` complete
    __syncthreads();     // all waves' staging visible; prev compute done
    if (it + 1 < nK) {   // overlap next-tile staging with this tile's WMMAs
      stageA(buf ^ 1, (it + 1) * TK_);
      stageB(buf ^ 1, (it + 1) * TK_);
    }

    // ---- load ALL fragments, then a clause of 8 WMMAs ----------------------
    // A (16x32 bf16): element e holds K = e + 8*half + (e>=8 ? 8 : 0)
    v8u au[2];
    #pragma unroll
    for (int mi = 0; mi < 2; ++mi) {
      const int arow = wv * 32 + mi * 16 + l15;
      #pragma unroll
      for (int p = 0; p < 8; ++p) {
        int k0 = 2 * p + 8 * lhalf + ((p >= 4) ? 8 : 0);
        au[mi][p] = *(const unsigned int*)&lsA[buf][arow * LDA_ + k0];
      }
    }
    // B (32x16 bf16): N = t*16 + (lane&15); element e holds K = e + 8*half + (e>=8 ? 8 : 0)
    v8u bu[4];
    #pragma unroll
    for (int t = 0; t < 4; ++t) {
      const int brow = t * 16 + l15;
      #pragma unroll
      for (int p = 0; p < 8; ++p) {
        int k0 = 2 * p + 8 * lhalf + ((p >= 4) ? 8 : 0);
        bu[t][p] = *(const unsigned int*)&lsB[buf][brow * LDB_ + k0];
      }
    }
    #pragma unroll
    for (int mi = 0; mi < 2; ++mi) {
      v16bf av = __builtin_bit_cast(v16bf, au[mi]);
      #pragma unroll
      for (int t = 0; t < 4; ++t) {
        acc[mi][t] = __builtin_amdgcn_wmma_f32_16x16x32_bf16(
            false, av, false, __builtin_bit_cast(v16bf, bu[t]),
            (short)0, acc[mi][t], false, false);
      }
    }
    buf ^= 1;
  }

  // ---- epilogue: C row = j + 8*half, col = lane&15 --------------------------
  const int colBase = nBase + l15;
  #pragma unroll
  for (int mi = 0; mi < 2; ++mi) {
    #pragma unroll
    for (int t = 0; t < 4; ++t) {
      const int col = colBase + t * 16;
      const float bv = bias[col];
      #pragma unroll
      for (int j = 0; j < 8; ++j) {
        const int m = mBase + wv * 32 + mi * 16 + 8 * lhalf + j;
        if (m < M) {
          float v = acc[mi][t][j] + bv;
          if (MODE == 0) {
            outBF[(size_t)m * N + col] = f2bf(tanhf(v));
          } else if (MODE == 1) {
            const int b = m / KR_;
            outF[(size_t)m * N + col] = tanhf(v + qt[(size_t)b * N + col]);
          } else {
            outF[(size_t)m * N + col] = v;
          }
        }
      }
    }
  }
}

// ---------------------------------------------------------------------------
// Fused softmax over R + weighted pool + residual. One thread per (b,h);
// adjacent threads -> adjacent h -> coalesced column reads (L2 resident).
// ---------------------------------------------------------------------------
__global__ __launch_bounds__(256) void softmax_pool_kernel(
    const float* __restrict__ ha, const unsigned short* __restrict__ v1,
    const float* __restrict__ vq, float* __restrict__ out) {
  const int t = blockIdx.x * 256 + threadIdx.x;   // 0 .. B*H-1
  const int h = t & (KH_ - 1);
  const int b = t >> 10;
  const float* col = ha + (size_t)b * KR_ * KH_ + h;
  const unsigned short* vcol = v1 + (size_t)b * KR_ * KH_ + h;

  float mx = -1e30f;
  for (int r = 0; r < KR_; ++r) mx = fmaxf(mx, col[(size_t)r * KH_]);
  float s = 0.f, w = 0.f;
  for (int r = 0; r < KR_; ++r) {
    float e = __expf(col[(size_t)r * KH_] - mx);
    s += e;
    w += e * bf2f(vcol[(size_t)r * KH_]);
  }
  out[t] = w / s + vq[t];
}

// ---------------------------------------------------------------------------
// Launch
// ---------------------------------------------------------------------------
extern "C" void kernel_launch(void* const* d_in, const int* in_sizes, int n_in,
                              void* d_out, int out_size, void* d_ws, size_t ws_size,
                              hipStream_t stream) {
  (void)in_sizes; (void)n_in; (void)out_size; (void)ws_size;
  const float* v_i = (const float*)d_in[0];   // [B,R,F]
  const float* v_q = (const float*)d_in[1];   // [B,H]
  const float* W1  = (const float*)d_in[2];   // [H,F]
  const float* b1  = (const float*)d_in[3];   // [H]
  const float* W2i = (const float*)d_in[4];   // [H,H]
  const float* b2i = (const float*)d_in[5];   // [H]
  const float* W2q = (const float*)d_in[6];   // [H,H]
  const float* b2q = (const float*)d_in[7];   // [H]
  float* out = (float*)d_out;                 // [B,H]

  char* ws = (char*)d_ws;
  size_t off = 0;
  unsigned short* ws_v1  = (unsigned short*)(ws + off); off += (size_t)KM_ * KH_ * 2;  // 25.7 MB
  float*          ws_ha  = (float*)(ws + off);          off += (size_t)KM_ * KH_ * 4;  // 51.4 MB
  float*          ws_qt  = (float*)(ws + off);          off += (size_t)KB_ * KH_ * 4;  // 256 KB
  unsigned short* ws_w1  = (unsigned short*)(ws + off); off += (size_t)KH_ * KF_ * 2;  // 4 MB
  unsigned short* ws_w2i = (unsigned short*)(ws + off); off += (size_t)KH_ * KH_ * 2;  // 2 MB
  unsigned short* ws_w2q = (unsigned short*)(ws + off); off += (size_t)KH_ * KH_ * 2;  // 2 MB

  // 1) weights -> bf16 once (L2-resident afterwards)
  cvt_f32_bf16_kernel<<<(KH_ * KF_ + 255) / 256, 256, 0, stream>>>(W1, ws_w1, KH_ * KF_);
  cvt_f32_bf16_kernel<<<(KH_ * KH_ + 255) / 256, 256, 0, stream>>>(W2i, ws_w2i, KH_ * KH_);
  cvt_f32_bf16_kernel<<<(KH_ * KH_ + 255) / 256, 256, 0, stream>>>(W2q, ws_w2q, KH_ * KH_);

  // 2) q_t = v_q @ W2q^T + b2q   (M=64, K=1024, N=1024)
  {
    dim3 grid(KH_ / TN_, (KB_ + TM_ - 1) / TM_);
    wmma_gemm_kernel<2, false><<<grid, NT_, 0, stream>>>(
        v_q, nullptr, ws_w2q, b2q, nullptr, ws_qt, nullptr, KB_, KH_, KH_);
  }
  // 3) v1 = tanh(v_i @ W1^T + b1) -> bf16   (M=12544, K=2048, N=1024)
  {
    dim3 grid(KH_ / TN_, KM_ / TM_);
    wmma_gemm_kernel<0, false><<<grid, NT_, 0, stream>>>(
        v_i, nullptr, ws_w1, b1, nullptr, nullptr, ws_v1, KM_, KF_, KH_);
  }
  // 4) h_a = tanh(v1 @ W2i^T + b2i + q_t[b]) -> f32   (M=12544, K=1024, N=1024)
  {
    dim3 grid(KH_ / TN_, KM_ / TM_);
    wmma_gemm_kernel<1, true><<<grid, NT_, 0, stream>>>(
        nullptr, ws_v1, ws_w2i, b2i, ws_qt, ws_ha, nullptr, KM_, KH_, KH_);
  }
  // 5) softmax over R + weighted pool + residual
  softmax_pool_kernel<<<(KB_ * KH_) / 256, 256, 0, stream>>>(ws_ha, ws_v1, v_q, out);
}
